// RNN_Branch_54193897341322
// MI455X (gfx1250) — matrix operand order repaired and checked by reading, not compiled, not run on hardware
//
#include <hip/hip_runtime.h>

typedef __attribute__((ext_vector_type(16))) __bf16 v16bf;
typedef __attribute__((ext_vector_type(8)))  float  v8f;

#define T_SEQ 130
#define EDIM  100
#define HDIM  256
#define G4    1024
#define VOC   5
#define BATCH 4096
#define TILEB 64

#define HROW   272              // padded h row pitch in bf16 elements (544B: +8 dwords -> bank rotation)
#define HROWB  544
#define HBYTES (TILEB * HROWB)  // 34816 per h buffer

// LDS layout (bytes)
#define LDS_B    0                        // B-frag double buffer: 2 bufs x (2 halves x 32KB) = 128KB
#define LDS_H0   131072                   // h buffer 0: 64 x 544B = 34816
#define LDS_H1   (LDS_H0 + HBYTES)        // 165888
#define LDS_P    (LDS_H1 + HBYTES)        // 200704 : P table [5][1024] f32 = 20KB
#define LDS_C    (LDS_P + 20480)          // 221184 : c state 8 waves x 8 chunks x 1KB = 64KB
#define LDS_TOT  (LDS_C + 65536)          // 286720 < 320KB

struct Frag32 { float4 lo, hi; };
static __device__ __forceinline__ v16bf as_v16bf(const Frag32& f) {
    return __builtin_bit_cast(v16bf, f);
}
static __device__ __forceinline__ v8f as_v8f(const Frag32& f) {
    return __builtin_bit_cast(v8f, f);
}
static __device__ __forceinline__ Frag32 as_frag(const v8f& v) {
    return __builtin_bit_cast(Frag32, v);
}
static __device__ __forceinline__ float sigm(float x) {
    return 1.f / (1.f + __expf(-x));
}

// Async global -> LDS copy (CDNA5, tracked by ASYNCcnt). dsaddr = LDS byte addr.
// No memory clobber: consumers only read after s_wait_asynccnt + barrier.
static __device__ __forceinline__ void async_ld_b128(unsigned ldsAddr, const void* g) {
    asm volatile("global_load_async_to_lds_b128 %0, %1, off"
                 :: "v"(ldsAddr), "v"((unsigned long long)(size_t)g));
}
static __device__ __forceinline__ void wait_async0() {
    asm volatile("s_wait_asynccnt 0x0" ::: "memory");
}

// ---------------------------------------------------------------------------
// Pack Wr [256][1024] f32 -> bf16 WMMA B-fragments, chunk-major, split lo/hi:
// [ch(16)][kt(8)][gate(4)][part(2)][lane(32)][8 halves]
// ---------------------------------------------------------------------------
__global__ void pack_wr_kernel(const float* __restrict__ Wr, __bf16* __restrict__ dst) {
    int idx  = blockIdx.x * 256 + threadIdx.x;   // 16384 threads
    int lane = idx & 31;
    int f    = idx >> 5;                          // 0..511 : ((ch*8 + kt)*4 + gate)
    int gate = f & 3;
    int kt   = (f >> 2) & 7;
    int ch   = f >> 5;
    int n    = (gate * 16 + ch) * 16 + (lane & 15);
    int kadd = (lane >> 4) * 8;
    __bf16* o = dst + (size_t)f * 512;            // 512 halfs = 1KB per fragment
#pragma unroll
    for (int h = 0; h < 8; ++h) {
        int k = kt * 32 + kadd + h;
        o[lane * 8 + h]       = (__bf16)Wr[k * G4 + n];          // lo part
        o[256 + lane * 8 + h] = (__bf16)Wr[(k + 16) * G4 + n];   // hi part
    }
}

// P[v][g] = sum_e emb[v][e] * Wk[e][g] + b[g]
__global__ void proj_p_kernel(const float* __restrict__ emb, const float* __restrict__ Wk,
                              const float* __restrict__ bias, float* __restrict__ P) {
    int idx = blockIdx.x * 256 + threadIdx.x;
    if (idx >= VOC * G4) return;
    int v = idx >> 10, g = idx & 1023;
    float s = bias[g];
    for (int e = 0; e < EDIM; ++e) s += emb[v * EDIM + e] * Wk[e * G4 + g];
    P[idx] = s;
}

// ---------------------------------------------------------------------------
// Main recurrent scan. grid = 128 (2 dirs x 64 batch tiles), 256 threads.
// wave = strip(4) x colhalf(2); wave owns 16 rows x 128 h-cols (8 chunks of 16).
// Explicit double-buffered fragment pipeline in the k-loop.
// ---------------------------------------------------------------------------
__global__ void __launch_bounds__(256, 1)
__attribute__((amdgpu_num_vgpr(256)))
lstm_scan_kernel(const int* __restrict__ ids,
                 const __bf16* __restrict__ wr_packed,  // [2][16][8][4][2][32][8]
                 const float* __restrict__ pproj,       // [2][5][1024]
                 float* __restrict__ hbuf)              // [4096][512]
{
    extern __shared__ char smem[];
    const unsigned smemBase = (unsigned)(size_t)(void*)smem;  // LDS byte base
    float* plds = (float*)(smem + LDS_P);

    const int tid  = threadIdx.x;
    const int lane = tid & 31;
    const int wave = tid >> 5;
    const int s    = wave & 3;    // row strip 0..3
    const int hv   = wave >> 2;   // column half 0..1
    const int dir  = blockIdx.x >> 6;
    const int btile = blockIdx.x & 63;
    const int rowBase = btile * TILEB;

    const char*  wrp = (const char*)(wr_packed + (size_t)dir * (16 * 8 * 4 * 32 * 16));
    const float* pp  = pproj + (size_t)dir * (VOC * G4);

    for (int i = tid; i < VOC * G4; i += 256) plds[i] = pp[i];
    {   // zero h buffer 0 (read buffer of t=0) and c state
        unsigned int* hz = (unsigned int*)(smem + LDS_H0);
        for (int i = tid; i < HBYTES / 4; i += 256) hz[i] = 0u;
        unsigned int* cz = (unsigned int*)(smem + LDS_C);
        for (int i = tid; i < 65536 / 4; i += 256) cz[i] = 0u;
    }

    // prologue: async-stage chunks {0, 8} into buffer 0
#pragma unroll
    for (int i = 0; i < 8; ++i) {
        unsigned o = tid * 16 + i * 4096;                  // 32KB per chunk
        async_ld_b128(smemBase + LDS_B + o,         wrp + o);
        async_ld_b128(smemBase + LDS_B + 32768 + o, wrp + (size_t)8 * 32768 + o);
    }
    wait_async0();
    __syncthreads();

    const int mrow  = s * 16 + (lane & 15);   // local row for A fragment
    const int kbase = (lane >> 4) * 8;        // A layout: low lanes K+0, high lanes K+8

    for (int t = 0; t < T_SEQ; ++t) {
        const int tt = dir ? (T_SEQ - 1 - t) : t;
        const char* hrd = smem + ((t & 1) ? LDS_H1 : LDS_H0);
        __bf16*     hwr = (__bf16*)(smem + ((t & 1) ? LDS_H0 : LDS_H1));

        int myid = 0;
        if (lane < 16)
            myid = ids[(size_t)(rowBase + s * 16 + lane) * T_SEQ + tt];

#pragma unroll
        for (int cc = 0; cc < 8; ++cc) {
            // async-stage next chunk pair into the other buffer
            {
                const int ncc = (cc + 1) & 7;
                const unsigned nb = (unsigned)((cc + 1) & 1) * 65536u;
                const char* srcA = wrp + (size_t)ncc * 32768;
                const char* srcB = wrp + (size_t)(8 + ncc) * 32768;
#pragma unroll
                for (int i = 0; i < 8; ++i) {
                    unsigned o = tid * 16 + i * 4096;
                    async_ld_b128(smemBase + LDS_B + nb + o,         srcA + o);
                    async_ld_b128(smemBase + LDS_B + nb + 32768 + o, srcB + o);
                }
            }

            // load c state (per-lane v8f, split 16B blocks: conflict-free)
            const char* cb = smem + LDS_C + (size_t)(wave * 8 + cc) * 1024;
            Frag32 cf;
            cf.lo = *(const float4*)(cb + lane * 16);
            cf.hi = *(const float4*)(cb + 512 + lane * 16);
            v8f cold = as_v8f(cf);

            const char* bbase = smem + LDS_B + (size_t)(cc & 1) * 65536 + (size_t)hv * 32768;
            v8f acc[4];
#pragma unroll
            for (int q = 0; q < 4; ++q)
#pragma unroll
                for (int r = 0; r < 8; ++r) acc[q][r] = 0.f;

            // ---- explicit double-buffered fragment pipeline over kt ----
            Frag32 abuf[2];
            Frag32 bbuf[2][4];
            {   // preload kt = 0
                const char* ap = hrd + mrow * HROWB + kbase * 2;
                abuf[0].lo = *(const float4*)(ap);
                abuf[0].hi = *(const float4*)(ap + 32);
#pragma unroll
                for (int gate = 0; gate < 4; ++gate) {
                    const char* fb = bbase + (size_t)gate * 1024;
                    bbuf[0][gate].lo = *(const float4*)(fb + lane * 16);
                    bbuf[0][gate].hi = *(const float4*)(fb + 512 + lane * 16);
                }
            }
#pragma unroll
            for (int kt = 0; kt < 8; ++kt) {
                const int cur = kt & 1;
                const int nxt = cur ^ 1;
                if (kt < 7) {   // issue next iteration's loads before this one's WMMAs
                    const char* ap = hrd + mrow * HROWB + ((kt + 1) * 32 + kbase) * 2;
                    abuf[nxt].lo = *(const float4*)(ap);
                    abuf[nxt].hi = *(const float4*)(ap + 32);
#pragma unroll
                    for (int gate = 0; gate < 4; ++gate) {
                        const char* fb = bbase + (size_t)((kt + 1) * 4 + gate) * 1024;
                        bbuf[nxt][gate].lo = *(const float4*)(fb + lane * 16);
                        bbuf[nxt][gate].hi = *(const float4*)(fb + 512 + lane * 16);
                    }
                }
                v16bf av = as_v16bf(abuf[cur]);
#pragma unroll
                for (int gate = 0; gate < 4; ++gate) {
                    acc[gate] = __builtin_amdgcn_wmma_f32_16x16x32_bf16(
                        false, av, false, as_v16bf(bbuf[cur][gate]), (short)0, acc[gate],
                        false, false);
                }
            }

            // fused gate math: i,f,g,o -> c,h   (z = acc + P[id][gate*256 + n])
            const int nh = (hv * 8 + cc) * 16 + (lane & 15);
            v8f cnew;
#pragma unroll
            for (int r = 0; r < 8; ++r) {
                const int rowSel = ((lane >> 4) << 3) + r;   // 0..15
                const int idv = __shfl(myid, rowSel, 32);
                const float* prow = plds + (size_t)idv * G4 + nh;
                float zi = acc[0][r] + prow[0];
                float zf = acc[1][r] + prow[256];
                float zg = acc[2][r] + prow[512];
                float zo = acc[3][r] + prow[768];
                float gi_ = sigm(zi);
                float gf  = sigm(zf);
                float gg  = fmaxf(zg, 0.f);
                float go  = sigm(zo);
                float cv = gf * cold[r] + gi_ * gg;
                cnew[r] = cv;
                float hval = go * fmaxf(cv, 0.f);
                const int m2 = s * 16 + rowSel;
                hwr[m2 * HROW + nh] = (__bf16)hval;
                if (t == T_SEQ - 1)
                    hbuf[(size_t)(rowBase + m2) * 512 + dir * HDIM + nh] = hval;
            }
            {   // store updated c state
                Frag32 cs = as_frag(cnew);
                *(float4*)((char*)cb + lane * 16)       = cs.lo;
                *(float4*)((char*)cb + 512 + lane * 16) = cs.hi;
            }

            wait_async0();
            __syncthreads();
        }
    }
}

// ---------------------------------------------------------------------------
// MLP head: out[b] = sigmoid(b2 + sum_k tanh(b1[k] + h[b,:] . W1[:,k]) * W2[k])
// ---------------------------------------------------------------------------
__global__ void head_kernel(const float* __restrict__ hbuf,
                            const float* __restrict__ W1, const float* __restrict__ b1,
                            const float* __restrict__ W2, const float* __restrict__ b2,
                            float* __restrict__ out) {
    int b = blockIdx.x * 256 + threadIdx.x;
    if (b >= BATCH) return;
    const float* h = hbuf + (size_t)b * 512;
    float acc = b2[0];
#pragma unroll
    for (int k = 0; k < 10; ++k) {
        float d = b1[k];
        for (int j = 0; j < 512; ++j) d += h[j] * W1[j * 10 + k];
        acc += tanhf(d) * W2[k];
    }
    out[b] = sigm(acc);
}

extern "C" void kernel_launch(void* const* d_in, const int* in_sizes, int n_in,
                              void* d_out, int out_size, void* d_ws, size_t ws_size,
                              hipStream_t stream) {
    const int*   ids  = (const int*)d_in[0];
    const float* emb  = (const float*)d_in[1];
    const float* Wk_f = (const float*)d_in[2];
    const float* Wr_f = (const float*)d_in[3];
    const float* b_f  = (const float*)d_in[4];
    const float* Wk_b = (const float*)d_in[5];
    const float* Wr_b = (const float*)d_in[6];
    const float* b_b  = (const float*)d_in[7];
    const float* W1   = (const float*)d_in[8];
    const float* b1   = (const float*)d_in[9];
    const float* W2   = (const float*)d_in[10];
    const float* b2   = (const float*)d_in[11];
    float* out = (float*)d_out;

    char* ws = (char*)d_ws;
    __bf16* wrPacked = (__bf16*)ws;                       // 2 x 262144 bf16 = 1MB
    float*  pproj    = (float*)(ws + 1048576);            // 2 x 5120 f32 = 40KB
    float*  hbuf     = (float*)(ws + 1048576 + 40960);    // 4096 x 512 f32 = 8MB

    pack_wr_kernel<<<64, 256, 0, stream>>>(Wr_f, wrPacked);
    pack_wr_kernel<<<64, 256, 0, stream>>>(Wr_b, wrPacked + 262144);
    proj_p_kernel<<<20, 256, 0, stream>>>(emb, Wk_f, b_f, pproj);
    proj_p_kernel<<<20, 256, 0, stream>>>(emb, Wk_b, b_b, pproj + VOC * G4);

    lstm_scan_kernel<<<128, 256, LDS_TOT, stream>>>(ids, wrPacked, pproj, hbuf);

    head_kernel<<<16, 256, 0, stream>>>(hbuf, W1, b1, W2, b2, out);
}
